// QNN_37477884624915
// MI455X (gfx1250) — matrix operand order repaired and checked by reading, not compiled, not run on hardware
//
#include <hip/hip_runtime.h>

typedef __attribute__((ext_vector_type(16))) _Float16 v16h;
typedef __attribute__((ext_vector_type(8)))  _Float16 v8h;
typedef __attribute__((ext_vector_type(8)))  float    v8f;

#define NQ   8
#define DIM  256          // 2^8 state dimension (GEMM K)
#define NN   512          // Re rows 0..255, Im rows 256..511 (GEMM N)
#define KT   8            // K tiles of 32
#define NT   32           // N tiles of 16
#define TILES_PER_BLK 8   // M tiles (16 rows each) per block
#define RSQRT2 0.70710678118654752440f

#define U_BYTES   (KT * NT * 32 * 16 * 2)           // 262144: U in B-frag layout
#define A_BYTES   (KT * 32 * 16 * 2)                // 8192:  A fragments
#define P_BYTES   (16 * 32 * 4)                     // 2048:  probability bins
#define SMEM_BYTES (U_BYTES + A_BYTES + P_BYTES)    // 272384 <= 320K WGP LDS

// ---------------------------------------------------------------------------
// WMMA f16 operand fragment layouts as packed here (contiguous k order, wave32):
//  A (16x32, M x K): lane = M%16 + 16*(Klocal>=16);
//    slot = Klocal & 15
//  B (32x16, K x N): lane = N%16 + 16*(Klocal>=16); slot = Klocal & 15
// ---------------------------------------------------------------------------
__device__ __forceinline__ int bfrag_index(int ks, int nt, int klocal, int nlocal) {
    int lane = nlocal + ((klocal & 16) ? 16 : 0);
    int slot = klocal & 15;
    return ((ks * NT + nt) * 32 + lane) * 16 + slot;
}

// ===========================================================================
// Kernel 1: build the fixed 256x256 complex entangling unitary U and store
// B = U^T (Re rows, then Im rows) in WMMA B-fragment layout in workspace.
// One block per basis-state column; state lives in LDS.
// ===========================================================================
__global__ __launch_bounds__(256) void qnn_build_U(const float* __restrict__ w,
                                                   _Float16* __restrict__ Ufrag) {
    __shared__ float st_re[DIM];
    __shared__ float st_im[DIM];
    const int tid = threadIdx.x;
    const int col = blockIdx.x;           // K index (input basis state)

    st_re[tid] = (tid == col) ? 1.0f : 0.0f;
    st_im[tid] = 0.0f;
    __syncthreads();

    const int ctrl[9] = {0, 1, 2, 3, 3, 4, 5, 6, 7};
    const int targ[9] = {1, 2, 3, 0, 4, 5, 6, 7, 3};

    for (int p = 0; p < 9; ++p) {
        const int c = ctrl[p], t = targ[p];
        const int tb = 7 - t, cb = 7 - c;      // wire q -> bit (7-q) of flat index
        const int tmask = 1 << tb;
        const float w0 = w[p * 5 + 0], w1 = w[p * 5 + 1], w2 = w[p * 5 + 2];
        const float w3 = w[p * 5 + 3], w4 = w[p * 5 + 4];

        // RZ(w0) on t: diag(e^{-i w/2}, e^{+i w/2})  (per-element)
        {
            float sh, ch; sincosf(0.5f * w0, &sh, &ch);
            float sgn = ((tid >> tb) & 1) ? sh : -sh;
            float re = st_re[tid], im = st_im[tid];
            st_re[tid] = re * ch - im * sgn;
            st_im[tid] = re * sgn + im * ch;
        }
        // CRZ(w1) c->t  (per-element, control bit == 1)
        if ((tid >> cb) & 1) {
            float sh, ch; sincosf(0.5f * w1, &sh, &ch);
            float sgn = ((tid >> tb) & 1) ? sh : -sh;
            float re = st_re[tid], im = st_im[tid];
            st_re[tid] = re * ch - im * sgn;
            st_im[tid] = re * sgn + im * ch;
        }
        __syncthreads();
        // RY(w2) on t  (mixes pairs; leader = target bit 0)
        if (!((tid >> tb) & 1)) {
            float sh, ch; sincosf(0.5f * w2, &sh, &ch);
            int i1 = tid | tmask;
            float r0 = st_re[tid], i0 = st_im[tid];
            float r1 = st_re[i1],  i1v = st_im[i1];
            st_re[tid] = ch * r0 - sh * r1;  st_im[tid] = ch * i0 - sh * i1v;
            st_re[i1]  = sh * r0 + ch * r1;  st_im[i1]  = sh * i0 + ch * i1v;
        }
        __syncthreads();
        // CX c->t  (swap pair where control==1)
        if (((tid >> cb) & 1) && !((tid >> tb) & 1)) {
            int i1 = tid | tmask;
            float r0 = st_re[tid], i0 = st_im[tid];
            st_re[tid] = st_re[i1]; st_im[tid] = st_im[i1];
            st_re[i1] = r0;         st_im[i1] = i0;
        }
        __syncthreads();
        // RY(w3) on t
        if (!((tid >> tb) & 1)) {
            float sh, ch; sincosf(0.5f * w3, &sh, &ch);
            int i1 = tid | tmask;
            float r0 = st_re[tid], i0 = st_im[tid];
            float r1 = st_re[i1],  i1v = st_im[i1];
            st_re[tid] = ch * r0 - sh * r1;  st_im[tid] = ch * i0 - sh * i1v;
            st_re[i1]  = sh * r0 + ch * r1;  st_im[i1]  = sh * i0 + ch * i1v;
        }
        __syncthreads();
        // CRZ(w4) c->t
        if ((tid >> cb) & 1) {
            float sh, ch; sincosf(0.5f * w4, &sh, &ch);
            float sgn = ((tid >> tb) & 1) ? sh : -sh;
            float re = st_re[tid], im = st_im[tid];
            st_re[tid] = re * ch - im * sgn;
            st_im[tid] = re * sgn + im * ch;
        }
        __syncthreads();
    }

    // Emit B[k=col][n] = U[n][col]:  Re at n = tid, Im at n = tid + 256.
    const int ks = col >> 5, klocal = col & 31;
    const int nt = tid >> 4, nlocal = tid & 15;
    Ufrag[bfrag_index(ks, nt,      klocal, nlocal)] = (_Float16)st_re[tid];
    Ufrag[bfrag_index(ks, nt + 16, klocal, nlocal)] = (_Float16)st_im[tid];
}

// ===========================================================================
// Kernel 2: 256 blocks x 8 M-tiles. Phase 0 pulls the whole 256 KB U image
// into LDS once per block with async global->LDS copies (ASYNCcnt path).
// Then per M-tile: embed 16 samples -> LDS A fragments; 8 waves x 4 N-tiles
// x 8 K-steps of v_wmma_f32_16x16x32_f16 over N=512 (Re|Im); fused |y|^2
// reduction (bin = n mod 32) and a full overwrite of the output tile.
// ===========================================================================
__global__ __launch_bounds__(256) void qnn_gemm(const float* __restrict__ x,
                                                const _Float16* __restrict__ Ufrag,
                                                float* __restrict__ out) {
    extern __shared__ char smem[];
    _Float16* ldsU = (_Float16*)smem;                       // 256 KB
    _Float16* ldsA = (_Float16*)(smem + U_BYTES);           // 8 KB
    float*    pacc = (float*)(smem + U_BYTES + A_BYTES);    // 2 KB

    const int tid  = threadIdx.x;
    const int wave = tid >> 5;
    const int lane = tid & 31;

    // ---- phase 0: U -> LDS, once per block, via async-to-LDS DMA ----
    {
        const unsigned ldsU_base = (unsigned)(uintptr_t)ldsU;  // LDS byte offset
        for (int i = 0; i < 64; ++i) {
            unsigned ch = (unsigned)(i * 256 + tid);           // 16384 x 16B chunks
            unsigned ldsoff = ldsU_base + ch * 16u;
            unsigned goff = ch * 16u;
            asm volatile("global_load_async_to_lds_b128 %0, %1, %2"
                         :: "v"(ldsoff), "v"(goff), "s"(Ufrag)
                         : "memory");
        }
        asm volatile("s_wait_asynccnt 0x0" ::: "memory");
    }
    __syncthreads();

    for (int mt = 0; mt < TILES_PER_BLK; ++mt) {
        const int blk = blockIdx.x * TILES_PER_BLK + mt;   // rows blk*16..+15

        pacc[tid] = 0.0f;
        pacc[tid + 256] = 0.0f;

        // ---- phase 1: product-state embedding -> A fragments in LDS ----
        {
            const int s = tid >> 4;             // sample-in-tile (M local)
            const int p = tid & 15;             // 16-wide K chunk
            const long b = (long)blk * 16 + s;
            float a0[NQ], a1[NQ];
#pragma unroll
            for (int q = 0; q < NQ; ++q) {
                float sh, ch; sincosf(0.5f * x[b * NQ + q], &sh, &ch);
                a0[q] = (ch - sh) * RSQRT2;     // |0> amplitude of RY*H|0>
                a1[q] = (ch + sh) * RSQRT2;     // |1> amplitude
            }
#pragma unroll
            for (int j = 0; j < 16; ++j) {
                int k = p * 16 + j;
                float prod = 1.0f;
#pragma unroll
                for (int q = 0; q < NQ; ++q)
                    prod *= ((k >> (7 - q)) & 1) ? a1[q] : a0[q];
                int ks = k >> 5, klocal = k & 31;
                int alane = s + ((klocal & 16) ? 16 : 0);
                int aslot = klocal & 15;
                ldsA[(ks * 32 + alane) * 16 + aslot] = (_Float16)prod;
            }
        }
        __syncthreads();

        // ---- phase 2: WMMA main loop, all operands LDS-resident ----
        v8f acc[4] = {v8f{}, v8f{}, v8f{}, v8f{}};
#pragma unroll
        for (int ks = 0; ks < KT; ++ks) {
            const _Float16* ap = ldsA + (ks * 32 + lane) * 16;
            v8h alo = *(const v8h*)(ap);
            v8h ahi = *(const v8h*)(ap + 8);
            v16h a = __builtin_shufflevector(alo, ahi,
                     0, 1, 2, 3, 4, 5, 6, 7, 8, 9, 10, 11, 12, 13, 14, 15);
            const _Float16* bp = ldsU + (((ks * NT) + wave * 4) * 32 + lane) * 16;
#pragma unroll
            for (int tt = 0; tt < 4; ++tt) {
                const _Float16* bpt = bp + tt * (32 * 16);
                v8h blo = *(const v8h*)(bpt);
                v8h bhi = *(const v8h*)(bpt + 8);
                v16h bb = __builtin_shufflevector(blo, bhi,
                          0, 1, 2, 3, 4, 5, 6, 7, 8, 9, 10, 11, 12, 13, 14, 15);
                acc[tt] = __builtin_amdgcn_wmma_f32_16x16x32_f16(
                    false, a, false, bb, (short)0, acc[tt], false, false);
            }
        }

        // ---- phase 3: fused epilogue  p[b, n&31] += y^2 ----
        {
            const int mhi = (lane >> 4) * 8;    // C-layout: row = g + 8*(lane/16)
            const int nlo = lane & 15;          // col = lane%16
#pragma unroll
            for (int tt = 0; tt < 4; ++tt) {
                int n = (wave * 4 + tt) * 16 + nlo;
                int colbin = n & 31;            // n and n+256 share the bin
#pragma unroll
                for (int g = 0; g < 8; ++g) {
                    float y = acc[tt][g];
                    atomicAdd(&pacc[(mhi + g) * 32 + colbin], y * y);
                }
            }
        }
        __syncthreads();

        // ---- phase 4: write 16x32 probabilities (full overwrite) ----
        out[((long)blk * 16 + (tid >> 5)) * 32 + (tid & 31)]     = pacc[tid];
        out[((long)blk * 16 + 8 + (tid >> 5)) * 32 + (tid & 31)] = pacc[tid + 256];
        __syncthreads();   // protect ldsA/pacc reuse next M-tile
    }
}

// ===========================================================================
extern "C" void kernel_launch(void* const* d_in, const int* in_sizes, int n_in,
                              void* d_out, int out_size, void* d_ws, size_t ws_size,
                              hipStream_t stream) {
    const float* x = (const float*)d_in[0];       // (B, 8) fp32
    const float* w = (const float*)d_in[1];       // (1, 45) fp32
    float* out = (float*)d_out;                    // (B, 32) fp32
    _Float16* Ufrag = (_Float16*)d_ws;             // 256 KB B-fragment image of U

    const int B = in_sizes[0] / NQ;                // 32768
    const int blocks = B / 16 / TILES_PER_BLK;     // 256
    qnn_build_U<<<DIM, 256, 0, stream>>>(w, Ufrag);
    qnn_gemm<<<blocks, 256, SMEM_BYTES, stream>>>(x, Ufrag, out);
}
